// DPAL_16690242912836
// MI455X (gfx1250) — matrix operand order repaired and checked by reading, not compiled, not run on hardware
//
#include <hip/hip_runtime.h>
#include <hip/hip_bf16.h>

// ---------------------------------------------------------------------------
// Fused attention for B=4, Lq=Lkv=4096, D=128 (fp32 in/out).
//   proj kernels:  q = (x1 Wq^T + bq) * log2(e)/sqrt(D)   -> f16 [B,L,D]
//                  k = (x2 Wk^T + bk)                      -> f16 [B,L,D]
//                  vT = (x3 Wv^T + bv)^T                   -> f16 [B,D,L]
//   attn kernel:   flash attention, v_wmma_f32_16x16x32_f16 for QK^T and PV,
//                  double-buffered GLOBAL_LOAD_ASYNC_TO_LDS staging.
// ---------------------------------------------------------------------------

typedef __attribute__((ext_vector_type(16))) _Float16 v16h;
typedef __attribute__((ext_vector_type(8)))  float    v8f;
typedef int v4i __attribute__((vector_size(16)));

#define BATCH 4
#define LQ    4096
#define LKV   4096
#define DDIM  128

#define KS_STRIDE 136   // 64-row K tile, padded (halves)
#define VS_STRIDE 72    // 128-row V^T tile, padded (halves)
#define PB_STRIDE 72    // P re-layout buffer, padded (halves)

#if __has_builtin(__builtin_amdgcn_global_load_async_to_lds_b128) && \
    __has_builtin(__builtin_amdgcn_s_wait_asynccnt)
#define USE_ASYNC 1
#else
#define USE_ASYNC 0
#endif

union FragAB {                 // 16x32 (A) or 32x16 (B) f16 fragment: 8 VGPRs
    v16h h;
    unsigned int u[8];
    uint4 q[2];
};

__device__ __forceinline__ v8f zero8() {
    v8f z = {0.f, 0.f, 0.f, 0.f, 0.f, 0.f, 0.f, 0.f};
    return z;
}

__device__ __forceinline__ void copy_b128(const _Float16* g, _Float16* l) {
#if USE_ASYNC
    __builtin_amdgcn_global_load_async_to_lds_b128(
        (__attribute__((address_space(1))) v4i*)g,
        (__attribute__((address_space(3))) v4i*)l, 0, 0);
#else
    *(uint4*)l = *(const uint4*)g;
#endif
}

__device__ __forceinline__ void wait_async_all() {
#if USE_ASYNC
    __builtin_amdgcn_s_wait_asynccnt(0);
#endif
}

// ---------------- projection: y[b][l][e] = dot(x[b][l],W[e]) + bias[e] ------
__global__ __launch_bounds__(256, 1)
void proj_rowmajor_kernel(const float* __restrict__ x, const float* __restrict__ W,
                          const float* __restrict__ bias, _Float16* __restrict__ y,
                          float scale)
{
    const int e    = threadIdx.x & 127;
    const int row  = blockIdx.x * 2 + (threadIdx.x >> 7);
    const int b    = blockIdx.y;
    const float4* xr = (const float4*)(x + ((size_t)b * LQ + row) * DDIM);
    const float4* wr = (const float4*)(W + (size_t)e * DDIM);
    float acc = 0.f;
#pragma unroll 8
    for (int i = 0; i < DDIM / 4; ++i) {
        float4 a = xr[i], w = wr[i];
        acc += a.x * w.x + a.y * w.y + a.z * w.z + a.w * w.w;
    }
    y[((size_t)b * LQ + row) * DDIM + e] = (_Float16)((acc + bias[e]) * scale);
}

// ------------- projection with transposed output: yT[b][e][l] ---------------
__global__ __launch_bounds__(256, 1)
void proj_transposed_kernel(const float* __restrict__ x, const float* __restrict__ W,
                            const float* __restrict__ bias, _Float16* __restrict__ yT)
{
    const int row = blockIdx.x * 128 + (threadIdx.x & 127);
    const int e   = blockIdx.y * 2 + (threadIdx.x >> 7);
    const int b   = blockIdx.z;
    const float4* xr = (const float4*)(x + ((size_t)b * LKV + row) * DDIM);
    const float4* wr = (const float4*)(W + (size_t)e * DDIM);
    float acc = 0.f;
#pragma unroll 8
    for (int i = 0; i < DDIM / 4; ++i) {
        float4 a = xr[i], w = wr[i];
        acc += a.x * w.x + a.y * w.y + a.z * w.z + a.w * w.w;
    }
    yT[((size_t)b * DDIM + e) * LKV + row] = (_Float16)(acc + bias[e]);
}

// --------------------------- flash attention --------------------------------
// grid: (LQ/128, BATCH), block: 128 threads = 4 waves.
// Each wave owns 32 query rows (2 strips of 16); KV tiled by 64, double-buffered.
__global__ __launch_bounds__(128, 1)
void attn_kernel(const _Float16* __restrict__ qf,
                 const _Float16* __restrict__ kf,
                 const _Float16* __restrict__ vTf,
                 float* __restrict__ out)
{
    extern __shared__ _Float16 smem[];
    _Float16* KsBase = smem;                         // [2][64][KS_STRIDE]
    _Float16* VsBase = KsBase + 2 * 64 * KS_STRIDE;  // [2][128][VS_STRIDE]
    _Float16* Pb     = VsBase + 2 * 128 * VS_STRIDE; // [4][32][PB_STRIDE]

    const int tid  = threadIdx.x;
    const int wave = tid >> 5;
    const int lane = tid & 31;
    const int g    = lane >> 4;     // half-wave group
    const int lr   = lane & 15;     // row (A) or column (B/C) index
    const int b    = blockIdx.y;
    const int q0   = blockIdx.x * 128 + wave * 32;

    // per-thread staging addresses (K: 64x128 halves; V^T: 128x64 halves)
    const int kRow = tid >> 1, kCh = (tid & 1) * 64;
    const _Float16* kSrcBase = kf  + ((size_t)b * LKV + kRow) * DDIM + kCh;
    const _Float16* vSrcBase = vTf + ((size_t)b * DDIM + tid) * LKV;

    // ---- load Q fragments in A-layout (scale already folded into qf) ----
    FragAB aQ[2][4];
#pragma unroll
    for (int p = 0; p < 2; ++p) {
        const _Float16* qrow = qf + ((size_t)b * LQ + q0 + p * 16 + lr) * DDIM;
#pragma unroll
        for (int t = 0; t < 4; ++t)
#pragma unroll
            for (int j = 0; j < 8; ++j) {
                const int kb = (j < 4) ? (g * 8 + 2 * j) : (16 + g * 8 + 2 * (j - 4));
                aQ[p][t].u[j] = *(const unsigned int*)(qrow + t * 32 + kb);
            }
    }

    v8f o[2][8];
    float m[2][8], l[2][8];
#pragma unroll
    for (int p = 0; p < 2; ++p)
#pragma unroll
        for (int r = 0; r < 8; ++r) {
            o[p][r] = zero8();
            m[p][r] = -1e30f;
            l[p][r] = 0.f;
        }

    // ---- prefetch tile 0 into buffer 0 ----
    {
        _Float16* kDst = KsBase + kRow * KS_STRIDE + kCh;
        _Float16* vDst = VsBase + tid * VS_STRIDE;
#pragma unroll
        for (int i = 0; i < 8; ++i) copy_b128(kSrcBase + i * 8, kDst + i * 8);
#pragma unroll
        for (int i = 0; i < 8; ++i) copy_b128(vSrcBase + i * 8, vDst + i * 8);
    }

    const int NTILES = LKV / 64;
    for (int it = 0; it < NTILES; ++it) {
        const int cur = it & 1;
        wait_async_all();       // own async writes for tile `it` landed in LDS
        __syncthreads();        // publish to all waves; prior reads retired

        // issue prefetch of tile it+1 into the other buffer (hides under WMMAs)
        if (it + 1 < NTILES) {
            const size_t koff = (size_t)(it + 1) * 64 * DDIM;
            const size_t voff = (size_t)(it + 1) * 64;
            _Float16* kDst = KsBase + (1 - cur) * 64 * KS_STRIDE + kRow * KS_STRIDE + kCh;
            _Float16* vDst = VsBase + (1 - cur) * 128 * VS_STRIDE + tid * VS_STRIDE;
#pragma unroll
            for (int i = 0; i < 8; ++i) copy_b128(kSrcBase + koff + i * 8, kDst + i * 8);
#pragma unroll
            for (int i = 0; i < 8; ++i) copy_b128(vSrcBase + voff + i * 8, vDst + i * 8);
        }

        const _Float16* Kcur = KsBase + cur * 64 * KS_STRIDE;
        const _Float16* Vcur = VsBase + cur * 128 * VS_STRIDE;

        // ---- S = Q K^T : 4 n-tiles x 4 k-chunks, B fragment shared by 2 strips
        v8f s[2][4];
#pragma unroll
        for (int p = 0; p < 2; ++p)
#pragma unroll
            for (int nt = 0; nt < 4; ++nt) s[p][nt] = zero8();
#pragma unroll
        for (int nt = 0; nt < 4; ++nt) {
#pragma unroll
            for (int t = 0; t < 4; ++t) {
                FragAB bK;
                const _Float16* kr = Kcur + (nt * 16 + lr) * KS_STRIDE + t * 32 + g * 8;
                bK.q[0] = *(const uint4*)(kr);
                bK.q[1] = *(const uint4*)(kr + 16);
                s[0][nt] = __builtin_amdgcn_wmma_f32_16x16x32_f16(
                    false, aQ[0][t].h, false, bK.h, (short)0, s[0][nt], false, false);
                s[1][nt] = __builtin_amdgcn_wmma_f32_16x16x32_f16(
                    false, aQ[1][t].h, false, bK.h, (short)0, s[1][nt], false, false);
            }
        }

        // ---- online softmax per strip (base-2; 16-lane butterfly) ----
#pragma unroll
        for (int p = 0; p < 2; ++p) {
            float mnew[8], alpha[8];
#pragma unroll
            for (int r = 0; r < 8; ++r) {
                float t = fmaxf(fmaxf(s[p][0][r], s[p][1][r]),
                                fmaxf(s[p][2][r], s[p][3][r]));
#pragma unroll
                for (int off = 8; off >= 1; off >>= 1)
                    t = fmaxf(t, __shfl_xor(t, off, 32));
                mnew[r]   = fmaxf(m[p][r], t);
                alpha[r]  = __builtin_exp2f(m[p][r] - mnew[r]);
                m[p][r]   = mnew[r];
            }
#pragma unroll
            for (int nt = 0; nt < 4; ++nt)
#pragma unroll
                for (int r = 0; r < 8; ++r)
                    s[p][nt][r] = __builtin_exp2f(s[p][nt][r] - mnew[r]);
#pragma unroll
            for (int r = 0; r < 8; ++r) {
                float t = s[p][0][r] + s[p][1][r] + s[p][2][r] + s[p][3][r];
#pragma unroll
                for (int off = 8; off >= 1; off >>= 1)
                    t += __shfl_xor(t, off, 32);
                l[p][r] = l[p][r] * alpha[r] + t;
            }
#pragma unroll
            for (int d = 0; d < 8; ++d)
#pragma unroll
                for (int r = 0; r < 8; ++r)
                    o[p][d][r] *= alpha[r];

            // ---- re-layout P (C-layout -> A-layout) through per-wave LDS ----
#pragma unroll
            for (int nt = 0; nt < 4; ++nt)
#pragma unroll
                for (int r = 0; r < 8; ++r)
                    Pb[(wave * 32 + p * 16 + r + 8 * g) * PB_STRIDE + nt * 16 + lr] =
                        (_Float16)s[p][nt][r];
        }
        __asm__ volatile("s_wait_dscnt 0" ::: "memory");

        FragAB aP[2][2];
#pragma unroll
        for (int p = 0; p < 2; ++p)
#pragma unroll
            for (int t = 0; t < 2; ++t) {
                const _Float16* pr = Pb + (wave * 32 + p * 16 + lr) * PB_STRIDE;
                aP[p][t].q[0] = *(const uint4*)(pr + t * 32 + g * 8);
                aP[p][t].q[1] = *(const uint4*)(pr + t * 32 + 16 + g * 8);
            }

        // ---- O += P V : 8 d-tiles x 2 k-chunks, B fragment shared by 2 strips
#pragma unroll
        for (int dt = 0; dt < 8; ++dt) {
#pragma unroll
            for (int t = 0; t < 2; ++t) {
                FragAB bV;
                const _Float16* vr = Vcur + (dt * 16 + lr) * VS_STRIDE + t * 32 + g * 8;
                bV.q[0] = *(const uint4*)(vr);
                bV.q[1] = *(const uint4*)(vr + 16);
                o[0][dt] = __builtin_amdgcn_wmma_f32_16x16x32_f16(
                    false, aP[0][t].h, false, bV.h, (short)0, o[0][dt], false, false);
                o[1][dt] = __builtin_amdgcn_wmma_f32_16x16x32_f16(
                    false, aP[1][t].h, false, bV.h, (short)0, o[1][dt], false, false);
            }
        }
    }

    // ---- normalize and write out ----
#pragma unroll
    for (int p = 0; p < 2; ++p) {
        float inv[8];
#pragma unroll
        for (int r = 0; r < 8; ++r) inv[r] = 1.0f / l[p][r];
#pragma unroll
        for (int dt = 0; dt < 8; ++dt)
#pragma unroll
            for (int r = 0; r < 8; ++r)
                out[((size_t)b * LQ + q0 + p * 16 + r + 8 * g) * DDIM + dt * 16 + lr] =
                    o[p][dt][r] * inv[r];
    }
}

// ---------------------------------------------------------------------------
extern "C" void kernel_launch(void* const* d_in, const int* in_sizes, int n_in,
                              void* d_out, int out_size, void* d_ws, size_t ws_size,
                              hipStream_t stream)
{
    const float* x1 = (const float*)d_in[0];
    const float* x2 = (const float*)d_in[1];
    const float* x3 = (const float*)d_in[2];
    const float* Wq = (const float*)d_in[3];
    const float* bq = (const float*)d_in[4];
    const float* Wk = (const float*)d_in[5];
    const float* bk = (const float*)d_in[6];
    const float* Wv = (const float*)d_in[7];
    const float* bv = (const float*)d_in[8];
    float* out = (float*)d_out;

    _Float16* qf = (_Float16*)d_ws;                       // [B][LQ][D]
    _Float16* kf = qf + (size_t)BATCH * LQ * DDIM;        // [B][LKV][D]
    _Float16* vT = kf + (size_t)BATCH * LKV * DDIM;       // [B][D][LKV]

    // fold 1/sqrt(D) and log2(e) into q so softmax uses native exp2
    const float qscale = 1.4426950408889634f / 11.313708498984761f;

    dim3 blk(256);
    proj_rowmajor_kernel<<<dim3(LQ / 2, BATCH), blk, 0, stream>>>(x1, Wq, bq, qf, qscale);
    proj_rowmajor_kernel<<<dim3(LKV / 2, BATCH), blk, 0, stream>>>(x2, Wk, bk, kf, 1.0f);
    proj_transposed_kernel<<<dim3(LKV / 128, DDIM / 2, BATCH), blk, 0, stream>>>(x3, Wv, bv, vT);

    const size_t smem_bytes =
        (size_t)(2 * 64 * KS_STRIDE + 2 * 128 * VS_STRIDE + 4 * 32 * PB_STRIDE) *
        sizeof(_Float16);  // ~88 KB (gfx1250 WGP: 320 KB LDS)
    attn_kernel<<<dim3(LQ / 128, BATCH), dim3(128), smem_bytes, stream>>>(qf, kf, vT, out);
}
